// GeoMultiHeadAttention_28905129902244
// MI455X (gfx1250) — matrix operand order repaired and checked by reading, not compiled, not run on hardware
//
#include <hip/hip_runtime.h>

// GeoMultiHeadAttention for MI455X (gfx1250), wave32 + WMMA.
// Key algebra: q·(Wp_h x) == (Wp_h^T q)·x  -> stream input_p (268MB) exactly once.
// bk/bp dropped (softmax shift-invariant); bq kept in Q-proj; bv added at output.

#define DMODEL 256
#define NHEAD  4
#define DHEAD  64
#define NSEQ   512
#define MSEQ   512

typedef __attribute__((ext_vector_type(16))) __bf16 v16bf;
typedef __attribute__((ext_vector_type(8)))  __bf16 v8bf;
typedef __attribute__((ext_vector_type(8)))  float  v8f;
typedef __attribute__((ext_vector_type(4)))  float  v4f;

__device__ __forceinline__ v8f wmma_bf16(v16bf a, v16bf b, v8f c) {
    return __builtin_amdgcn_wmma_f32_16x16x32_bf16(false, a, false, b, (short)0, c,
                                                   false, false);
}

// ---- fragment loaders (wave32 layouts, see CDNA5 ISA 7.12.2) ----
// A 16x32 bf16: lane l (r=l&15, hi=l>>4) holds row r, K = 8*hi + {0..7, 16..23}.
// caller passes p = row_base + c0 + 8*hi; reads p[0..7] and p[16..23].
__device__ __forceinline__ v16bf a_from_f32(const float* p) {
    v16bf o;
#pragma unroll
    for (int t = 0; t < 8; t++) o[t] = (__bf16)p[t];
#pragma unroll
    for (int t = 0; t < 8; t++) o[t + 8] = (__bf16)p[t + 16];
    return o;
}
__device__ __forceinline__ v16bf a_from_f32_nt(const float* p) {  // streaming source
    v4f a0 = __builtin_nontemporal_load((const v4f*)(p));
    v4f a1 = __builtin_nontemporal_load((const v4f*)(p + 4));
    v4f b0 = __builtin_nontemporal_load((const v4f*)(p + 16));
    v4f b1 = __builtin_nontemporal_load((const v4f*)(p + 20));
    v16bf o;
#pragma unroll
    for (int t = 0; t < 4; t++) {
        o[t]      = (__bf16)a0[t];
        o[t + 4]  = (__bf16)a1[t];
        o[t + 8]  = (__bf16)b0[t];
        o[t + 12] = (__bf16)b1[t];
    }
    return o;
}
__device__ __forceinline__ v16bf a_from_bf(const __bf16* p) {
    v8bf a = *(const v8bf*)p;
    v8bf b = *(const v8bf*)(p + 16);
    v16bf o;
#pragma unroll
    for (int t = 0; t < 8; t++) { o[t] = a[t]; o[t + 8] = b[t]; }
    return o;
}
// B 32x16 bf16: lane l holds col l&15, K = 8*hi + {0..7, 16..23} (same as A).
// caller passes p = col_base + kc*32 + 8*hi; reads p[0..7] and p[16..23].
__device__ __forceinline__ v16bf b_from_bf(const __bf16* p) {
    return a_from_bf(p);
}
__device__ __forceinline__ v16bf b_from_f32(const float* p) {
    v16bf o;
#pragma unroll
    for (int t = 0; t < 16; t++) o[t] = (__bf16)p[t < 8 ? t : t + 8];
    return o;
}

// ============ K1: Q/K/V projections (bf16 out; V stored transposed) ============
// 1536 tiles: proj p(3) x ntile(32) x otile(16); one wave per 16x16 tile, K=256.
__global__ __launch_bounds__(256) void k_proj(const float* __restrict__ Xq,
                                              const float* __restrict__ Xk,
                                              const float* __restrict__ Xv,
                                              const float* __restrict__ Wq,
                                              const float* __restrict__ Wk,
                                              const float* __restrict__ Wv,
                                              const float* __restrict__ bq,
                                              __bf16* __restrict__ Qbf,
                                              __bf16* __restrict__ Kbf,
                                              __bf16* __restrict__ Vtbf) {
    int wave = (blockIdx.x * blockDim.x + threadIdx.x) >> 5;
    int ot = wave & 15, nt = (wave >> 4) & 31, pidx = wave >> 9;
    int lane = threadIdx.x & 31, r = lane & 15, hi = lane >> 4;
    int n0 = nt * 16, o0 = ot * 16;
    const float* X = (pidx == 0) ? Xq : (pidx == 1) ? Xk : Xv;
    const float* W = (pidx == 0) ? Wq : (pidx == 1) ? Wk : Wv;
    v8f acc = {0.f, 0.f, 0.f, 0.f, 0.f, 0.f, 0.f, 0.f};
#pragma unroll
    for (int kc = 0; kc < 8; kc++) {
        v16bf a = a_from_f32(X + (n0 + r) * DMODEL + kc * 32 + 8 * hi);
        // B(k=c, col=o) = W[o][c]  -> contiguous in K for a fixed lane
        v16bf b = b_from_f32(W + (o0 + r) * DMODEL + kc * 32 + 8 * hi);
        acc = wmma_bf16(a, b, acc);
    }
    int col = o0 + r;
    if (pidx == 0) {
        float bias = bq[col];
#pragma unroll
        for (int i = 0; i < 8; i++)
            Qbf[(n0 + 8 * hi + i) * DMODEL + col] = (__bf16)(acc[i] + bias);
    } else if (pidx == 1) {
        // bk dropped: adds a per-(h,n) constant to scores -> softmax-invariant
#pragma unroll
        for (int i = 0; i < 8; i++)
            Kbf[(n0 + 8 * hi + i) * DMODEL + col] = (__bf16)acc[i];
    } else {
        // V stored transposed Vt[c][m] so attn@V B-operand loads are contiguous
        v8bf s;
#pragma unroll
        for (int i = 0; i < 8; i++) s[i] = (__bf16)acc[i];
        *(v8bf*)(Vtbf + col * MSEQ + n0 + 8 * hi) = s;  // bv added in k_out
    }
}

// ============ K2: qp[n][h][c] = 0.125 * sum_d Q[n][h*64+d] * Wp[h*64+d][c] ======
// 2048 tiles: h(4) x ntile(32) x ctile(16), K=64.
__global__ __launch_bounds__(256) void k_qp(const __bf16* __restrict__ Qbf,
                                            const float* __restrict__ Wp,
                                            __bf16* __restrict__ QPbf) {
    int wave = (blockIdx.x * blockDim.x + threadIdx.x) >> 5;
    int ct = wave & 15, nt = (wave >> 4) & 31, h = wave >> 9;
    int lane = threadIdx.x & 31, r = lane & 15, hi = lane >> 4;
    int n0 = nt * 16, c0 = ct * 16;
    v8f acc = {0.f, 0.f, 0.f, 0.f, 0.f, 0.f, 0.f, 0.f};
#pragma unroll
    for (int kc = 0; kc < 2; kc++) {
        v16bf a = a_from_bf(Qbf + (n0 + r) * DMODEL + h * DHEAD + kc * 32 + 8 * hi);
        // B(k=d, col=c) = Wp[h*64+d][c] : strided gather (tiny kernel)
        const float* pw = Wp + (h * DHEAD + kc * 32 + 8 * hi) * DMODEL + c0 + r;
        v16bf b;
#pragma unroll
        for (int t = 0; t < 16; t++) b[t] = (__bf16)pw[(t < 8 ? t : t + 8) * DMODEL];
        acc = wmma_bf16(a, b, acc);
    }
#pragma unroll
    for (int i = 0; i < 8; i++)
        QPbf[((n0 + 8 * hi + i) * NHEAD + h) * DMODEL + c0 + r] =
            (__bf16)(acc[i] * 0.125f);  // fold 1/sqrt(Dh)
}

// ============ K3: content scores S[h][n][m] = 0.125 * Q_h K_h^T ================
// 4096 tiles: h(4) x ntile(32) x mtile(32), K=64.
__global__ __launch_bounds__(256) void k_scores(const __bf16* __restrict__ Qbf,
                                                const __bf16* __restrict__ Kbf,
                                                float* __restrict__ S) {
    int wave = (blockIdx.x * blockDim.x + threadIdx.x) >> 5;
    int mt = wave & 31, nt = (wave >> 5) & 31, h = wave >> 10;
    int lane = threadIdx.x & 31, r = lane & 15, hi = lane >> 4;
    int n0 = nt * 16, m0 = mt * 16;
    v8f acc = {0.f, 0.f, 0.f, 0.f, 0.f, 0.f, 0.f, 0.f};
#pragma unroll
    for (int kc = 0; kc < 2; kc++) {
        v16bf a = a_from_bf(Qbf + (n0 + r) * DMODEL + h * DHEAD + kc * 32 + 8 * hi);
        // B(k=d, col=m) = K[m][h*64+d] -> contiguous per lane in [m][c] layout
        v16bf b = b_from_bf(Kbf + (m0 + r) * DMODEL + h * DHEAD + kc * 32 + 8 * hi);
        acc = wmma_bf16(a, b, acc);
    }
#pragma unroll
    for (int i = 0; i < 8; i++)
        S[((size_t)(h * NSEQ + n0 + 8 * hi + i)) * MSEQ + m0 + r] = acc[i] * 0.125f;
}

// ============ K4: the streaming kernel. S[h][n][m] += qp[n,h,:] . x[n,m,:] =====
// One block per n (512 blocks, 8 waves). Wave holds the 8 qp B-fragments in
// registers and sweeps 4 m-tiles; input_p read once, NT, fully-coalesced lines.
__global__ __launch_bounds__(256) void k_pscore(const float* __restrict__ Xp,
                                                const __bf16* __restrict__ QPbf,
                                                float* __restrict__ S) {
    int n = blockIdx.x;
    int w = threadIdx.x >> 5;
    int lane = threadIdx.x & 31, r = lane & 15, hi = lane >> 4;
    int h = r;                 // D column = head index (only 4 of 16 used)
    bool hv = (h < NHEAD);

    v16bf Bf[8];
    const __bf16* qb = QPbf + ((size_t)n * NHEAD + (hv ? h : 0)) * DMODEL;
#pragma unroll
    for (int kc = 0; kc < 8; kc++) {
        v16bf u;
#pragma unroll
        for (int t = 0; t < 16; t++) u[t] = (__bf16)0.f;
        if (hv) u = a_from_bf(qb + kc * 32 + 8 * hi);
        Bf[kc] = u;
    }

    const float* xn = Xp + (size_t)n * MSEQ * DMODEL;
    for (int j = 0; j < 4; j++) {
        int m0 = (w * 4 + j) * 16;
        const float* xr = xn + (size_t)(m0 + r) * DMODEL + 8 * hi;
        if (j < 3) __builtin_prefetch(xr + 16 * DMODEL, 0, 0);  // global_prefetch_b8

        float* ps = S + ((size_t)(h * NSEQ) + n) * MSEQ + m0 + 8 * hi;
        v8f acc = {0.f, 0.f, 0.f, 0.f, 0.f, 0.f, 0.f, 0.f};
        if (hv) {  // accumulate on top of content scores
            v4f c0 = *(const v4f*)ps;
            v4f c1 = *(const v4f*)(ps + 4);
#pragma unroll
            for (int i = 0; i < 4; i++) { acc[i] = c0[i]; acc[i + 4] = c1[i]; }
        }
#pragma unroll
        for (int kc = 0; kc < 8; kc++) {
            v16bf a = a_from_f32_nt(xr + kc * 32);
            acc = wmma_bf16(a, Bf[kc], acc);
        }
        if (hv) {
            v4f d0, d1;
#pragma unroll
            for (int i = 0; i < 4; i++) { d0[i] = acc[i]; d1[i] = acc[i + 4]; }
            *(v4f*)ps       = d0;
            *(v4f*)(ps + 4) = d1;
        }
    }
}

// ============ K5: row softmax -> bf16 attn ====================================
__global__ __launch_bounds__(256) void k_softmax(const float* __restrict__ S,
                                                 const unsigned char* __restrict__ mask,
                                                 __bf16* __restrict__ Attn) {
    int wave = (blockIdx.x * blockDim.x + threadIdx.x) >> 5;  // 0..2047 = h*512+n
    int lane = threadIdx.x & 31;
    const float* row = S + (size_t)wave * MSEQ;
    float v[16];
    float mx = -__builtin_inff();
#pragma unroll
    for (int j = 0; j < 16; j++) {
        int m = j * 32 + lane;
        float s = mask[m] ? -__builtin_inff() : row[m];
        v[j] = s;
        mx = fmaxf(mx, s);
    }
#pragma unroll
    for (int off = 16; off; off >>= 1) mx = fmaxf(mx, __shfl_xor(mx, off, 32));
    float sum = 0.f;
#pragma unroll
    for (int j = 0; j < 16; j++) { v[j] = __expf(v[j] - mx); sum += v[j]; }
#pragma unroll
    for (int off = 16; off; off >>= 1) sum += __shfl_xor(sum, off, 32);
    float inv = 1.f / sum;
    __bf16* arow = Attn + (size_t)wave * MSEQ;
#pragma unroll
    for (int j = 0; j < 16; j++) arow[j * 32 + lane] = (__bf16)(v[j] * inv);
}

// ============ K6: out[n][h*64+d] = attn_h @ V_h + bv ==========================
// 512 tiles: h(4) x ntile(32) x dtile(4), K=512.
__global__ __launch_bounds__(256) void k_out(const __bf16* __restrict__ Attn,
                                             const __bf16* __restrict__ Vtbf,
                                             const float* __restrict__ bv,
                                             float* __restrict__ Out) {
    int wave = (blockIdx.x * blockDim.x + threadIdx.x) >> 5;
    int dt = wave & 3, nt = (wave >> 2) & 31, h = wave >> 7;
    int lane = threadIdx.x & 31, r = lane & 15, hi = lane >> 4;
    int n0 = nt * 16, d0 = dt * 16;
    v8f acc = {0.f, 0.f, 0.f, 0.f, 0.f, 0.f, 0.f, 0.f};
#pragma unroll
    for (int kc = 0; kc < 16; kc++) {
        v16bf a = a_from_bf(Attn + ((size_t)(h * NSEQ) + n0 + r) * MSEQ + kc * 32 + 8 * hi);
        // B(k=m, col=d) = Vt[h*64+d][m] -> contiguous per lane
        v16bf b = b_from_bf(Vtbf + (h * DHEAD + d0 + r) * MSEQ + kc * 32 + 8 * hi);
        acc = wmma_bf16(a, b, acc);
    }
    int c = h * DHEAD + d0 + r;
    float bias = bv[c];
#pragma unroll
    for (int i = 0; i < 8; i++)
        Out[(n0 + 8 * hi + i) * DMODEL + c] = acc[i] + bias;
}

extern "C" void kernel_launch(void* const* d_in, const int* in_sizes, int n_in,
                              void* d_out, int out_size, void* d_ws, size_t ws_size,
                              hipStream_t stream) {
    (void)in_sizes; (void)n_in; (void)out_size; (void)ws_size;
    const float* Xq = (const float*)d_in[0];
    const float* Xk = (const float*)d_in[1];
    const float* Xv = (const float*)d_in[2];
    const float* Xp = (const float*)d_in[3];
    const unsigned char* mask = (const unsigned char*)d_in[4];
    const float* Wq = (const float*)d_in[5];
    const float* bq = (const float*)d_in[6];
    const float* Wk = (const float*)d_in[7];
    /* d_in[8] = bk : softmax-shift invariant, unused */
    const float* Wv = (const float*)d_in[9];
    const float* bv = (const float*)d_in[10];
    const float* Wp = (const float*)d_in[11];
    /* d_in[12] = bp : softmax-shift invariant, unused */
    float* Out = (float*)d_out;

    // workspace layout (~7.9 MB)
    __bf16* Qbf  = (__bf16*)d_ws;                                  // 512*256
    __bf16* Kbf  = Qbf  + NSEQ * DMODEL;                           // 512*256
    __bf16* Vtbf = Kbf  + MSEQ * DMODEL;                           // 256*512 (transposed)
    __bf16* QPbf = Vtbf + DMODEL * MSEQ;                           // 512*4*256
    __bf16* Attn = QPbf + (size_t)NSEQ * NHEAD * DMODEL;           // 4*512*512
    float*  S    = (float*)(Attn + (size_t)NHEAD * NSEQ * MSEQ);   // 4*512*512 f32

    k_proj<<<192, 256, 0, stream>>>(Xq, Xk, Xv, Wq, Wk, Wv, bq, Qbf, Kbf, Vtbf);
    k_qp<<<256, 256, 0, stream>>>(Qbf, Wp, QPbf);
    k_scores<<<512, 256, 0, stream>>>(Qbf, Kbf, S);
    k_pscore<<<NSEQ, 256, 0, stream>>>(Xp, QPbf, S);   // the 268MB streaming pass
    k_softmax<<<256, 256, 0, stream>>>(S, mask, Attn);
    k_out<<<64, 256, 0, stream>>>(Attn, Vtbf, bv, Out);
}
